// Diffusion_38603166057072
// MI455X (gfx1250) — matrix operand order repaired and checked by reading, not compiled, not run on hardware
//
#include <hip/hip_runtime.h>
#include <hip/hip_bf16.h>
#include <stdint.h>

// Problem constants (match reference)
constexpr int Bn = 8, Tn = 16, Ln = 128, Rn = 1024, En = 5, Fn = 512;
constexpr int LT = Ln / 16;   // 8 l-tiles
constexpr int RT = Rn / 16;   // 64 r-tiles

typedef __bf16 v16bf __attribute__((ext_vector_type(16)));
typedef __bf16 v8bf  __attribute__((ext_vector_type(8)));
typedef float  v8f   __attribute__((ext_vector_type(8)));

// ---- bf16 helpers (RNE, on bit patterns) -------------------------------
__device__ __forceinline__ uint16_t f32_to_bf16_bits(float x) {
  union { float f; uint32_t u; } v; v.f = x;
  uint32_t r = v.u + 0x7FFFu + ((v.u >> 16) & 1u);
  return (uint16_t)(r >> 16);
}
__device__ __forceinline__ float bf16_bits_to_f32(uint16_t s) {
  union { uint32_t u; float f; } v; v.u = ((uint32_t)s) << 16;
  return v.f;
}

// ---- Phase 0: split fp32 features into bf16 hi/lo planes ---------------
// hi = RNE(x), lo = RNE(x - hi): x ~= hi + lo to ~2^-17 relative.
__global__ void pack_kernel(const float* __restrict__ src,
                            uint16_t* __restrict__ hi,
                            uint16_t* __restrict__ lo, int n) {
  for (int i = blockIdx.x * blockDim.x + threadIdx.x; i < n;
       i += gridDim.x * blockDim.x) {
    float x = src[i];
    uint16_t h = f32_to_bf16_bits(x);
    hi[i] = h;
    lo[i] = f32_to_bf16_bits(x - bf16_bits_to_f32(h));
  }
}

// ---- Phase 1: QR(pre_rot) + rigid transform ---------------------------
// new_lig[b,t,l,:] = Q(pre_rot[b,t]) @ lig_coord[b,l,:] + trans[b,t,:]
__global__ void prep_kernel(const float* __restrict__ lig_coord,
                            const float* __restrict__ pre_rot,
                            const float* __restrict__ trans,
                            float* __restrict__ new_lig) {
  const int bt = blockIdx.x;          // 0 .. B*T-1
  const int b  = bt >> 4;
  const int l  = threadIdx.x;         // 0 .. L-1

  // Householder QR of 3x3 (LAPACK sign convention: alpha = -sign(x0)*||x||)
  float Rm[3][3], Q[3][3];
  const float* Ap = pre_rot + bt * 9;
#pragma unroll
  for (int i = 0; i < 3; ++i)
#pragma unroll
    for (int j = 0; j < 3; ++j) {
      Rm[i][j] = Ap[i * 3 + j];
      Q[i][j]  = (i == j) ? 1.0f : 0.0f;
    }
#pragma unroll
  for (int k = 0; k < 2; ++k) {
    float nrm2 = 0.0f;
    for (int i = k; i < 3; ++i) nrm2 += Rm[i][k] * Rm[i][k];
    float nrm = sqrtf(nrm2);
    float alpha = (Rm[k][k] >= 0.0f) ? -nrm : nrm;
    float v[3] = {0.0f, 0.0f, 0.0f};
    for (int i = k; i < 3; ++i) v[i] = Rm[i][k];
    v[k] -= alpha;
    float vtv = v[0] * v[0] + v[1] * v[1] + v[2] * v[2];
    if (vtv > 1e-30f) {
      float c = 2.0f / vtv;
#pragma unroll
      for (int j = 0; j < 3; ++j) {            // R = (I - c v v^T) R
        float s = v[0] * Rm[0][j] + v[1] * Rm[1][j] + v[2] * Rm[2][j];
#pragma unroll
        for (int i = 0; i < 3; ++i) Rm[i][j] -= c * s * v[i];
      }
#pragma unroll
      for (int i = 0; i < 3; ++i) {            // Q = Q (I - c v v^T)
        float s = Q[i][0] * v[0] + Q[i][1] * v[1] + Q[i][2] * v[2];
#pragma unroll
        for (int j = 0; j < 3; ++j) Q[i][j] -= c * s * v[j];
      }
    }
  }

  const float* xc = lig_coord + (b * Ln + l) * 3;
  const float* tr = trans + bt * 3;
  float x = xc[0], y = xc[1], z = xc[2];
  float* o = new_lig + (bt * Ln + l) * 3;
  o[0] = Q[0][0] * x + Q[0][1] * y + Q[0][2] * z + tr[0];
  o[1] = Q[1][0] * x + Q[1][1] * y + Q[1][2] * z + tr[1];
  o[2] = Q[2][0] * x + Q[2][1] * y + Q[2][2] * z + tr[2];
}

// Load a WMMA operand half (8 consecutive bf16 = one b128) then splice.
__device__ __forceinline__ v16bf load_op16(const uint16_t* p0,
                                           const uint16_t* p1) {
  v8bf a = *reinterpret_cast<const v8bf*>(p0);
  v8bf b = *reinterpret_cast<const v8bf*>(p1);
  return __builtin_shufflevector(a, b, 0, 1, 2, 3, 4, 5, 6, 7,
                                 8, 9, 10, 11, 12, 13, 14, 15);
}

// ---- Phase 2: WMMA feature GEMM + distance contraction -----------------
// One wave owns one 16x16 (l,r) tile of one graph b.
// grid = B * LT * (RT/8) blocks, 256 threads (8 waves; wave w -> rt = rtb*8+w)
__global__ void __launch_bounds__(256)
main_kernel(const uint16_t* __restrict__ lig_hi,  // [B,L,E,F] bf16 plane
            const uint16_t* __restrict__ lig_lo,
            const uint16_t* __restrict__ rec_hi,  // [B,R,E,F] bf16 plane
            const uint16_t* __restrict__ rec_lo,
            const float*    __restrict__ rec_coord,  // [B,R,3]
            const int*      __restrict__ lig_counts, // [B]
            const int*      __restrict__ rec_counts, // [B]
            const float*    __restrict__ new_lig,    // [B,T,L,3]  (ws)
            float*          __restrict__ partial) {  // [B,T,LT,RT] (ws)
  const int blk  = blockIdx.x;        // 0 .. B*LT*(RT/8)-1
  const int b    = blk >> 6;
  const int lt   = (blk >> 3) & 7;
  const int rtb  = blk & 7;
  const int wave = threadIdx.x >> 5;
  const int lane = threadIdx.x & 31;
  const int rt   = rtb * 8 + wave;    // 0 .. 63

  // Stage the pose tile: positions of the 16 l-rows for all T poses.
  __shared__ float s_pos[Tn * 16 * 3];          // 3 KB
  {
    int tid = threadIdx.x;                       // exactly 256 (t, lm) pairs
    int t  = tid >> 4;
    int lm = tid & 15;
    const float* p = new_lig + (((b * Tn + t) * Ln) + lt * 16 + lm) * 3;
    s_pos[tid * 3 + 0] = p[0];
    s_pos[tid * 3 + 1] = p[1];
    s_pos[tid * 3 + 2] = p[2];
  }
  __syncthreads();

  const int m16 = lane & 15;      // A row (M) / B col (N) owned by this lane
  const int kh  = lane >> 4;      // which K half
  const int kh8 = kh * 8, kh16 = kh * 16;

  const int lA = lt * 16 + m16;   // A-matrix row (ligand atom)
  const int rB = rt * 16 + m16;   // B-matrix col (receptor atom)

  // 5 accumulator tiles (one per exponent channel), split-bf16 fp32 GEMM:
  // (ahi+alo)(bhi+blo) ~= ahi*bhi + ahi*blo + alo*bhi  (3 bf16 WMMAs)
  v8f accs[En];
#pragma unroll
  for (int e = 0; e < En; ++e) {
    v8f acc = {0.f, 0.f, 0.f, 0.f, 0.f, 0.f, 0.f, 0.f};
    const int aOff = ((b * Ln + lA) * En + e) * Fn;
    const int bOff = ((b * Rn + rB) * En + e) * Fn;
    const uint16_t* aH = lig_hi + aOff;
    const uint16_t* aL = lig_lo + aOff;
    const uint16_t* bH = rec_hi + bOff;
    const uint16_t* bL = rec_lo + bOff;
    for (int f0 = 0; f0 < Fn; f0 += 32) {
      // A 16x32 bf16 layout: elems 0..7 -> K = kh*8+j, 8..15 -> K = 16+kh*8+j
      v16bf ahi = load_op16(aH + f0 + kh8, aH + f0 + 16 + kh8);
      v16bf alo = load_op16(aL + f0 + kh8, aL + f0 + 16 + kh8);
      // B 32x16 bf16 layout: lane owns col N, elems 0..7 -> K = kh*8+j, 8..15 -> K = 16+kh*8+j
      v16bf bhi = load_op16(bH + f0 + kh8, bH + f0 + 16 + kh8);
      v16bf blo = load_op16(bL + f0 + kh8, bL + f0 + 16 + kh8);
      acc = __builtin_amdgcn_wmma_f32_16x16x32_bf16(false, ahi, false, bhi,
                                                    (short)0, acc, false, false);
      acc = __builtin_amdgcn_wmma_f32_16x16x32_bf16(false, ahi, false, blo,
                                                    (short)0, acc, false, false);
      acc = __builtin_amdgcn_wmma_f32_16x16x32_bf16(false, alo, false, bhi,
                                                    (short)0, acc, false, false);
    }
    accs[e] = acc;
  }

  // Pair mask: C element j maps to (M = kh*8 + j, N = lane&15)
  const int nlig = lig_counts[b];
  const int nrec = rec_counts[b];
  const bool rOk = rB < nrec;
  bool vmask[8];
#pragma unroll
  for (int j = 0; j < 8; ++j) {
    int lIdx = lt * 16 + kh8 + j;
    vmask[j] = rOk && (lIdx < nlig);
    if (!vmask[j]) {
#pragma unroll
      for (int e = 0; e < En; ++e) accs[e][j] = 0.0f;
    }
  }

  const float rx = rec_coord[(b * Rn + rB) * 3 + 0];
  const float ry = rec_coord[(b * Rn + rB) * 3 + 1];
  const float rz = rec_coord[(b * Rn + rB) * 3 + 2];

  // Distance powers + contraction per pose, then wave reduce.
  float usum[Tn];
#pragma unroll
  for (int t = 0; t < Tn; ++t) {
    float s = 0.0f;
#pragma unroll
    for (int j = 0; j < 8; ++j) {
      int m = kh8 + j;
      const float* p = &s_pos[(t * 16 + m) * 3];
      float dx = p[0] - rx, dy = p[1] - ry, dz = p[2] - rz;
      float d2 = dx * dx + dy * dy + dz * dz;
      d2 = vmask[j] ? d2 : 1.0f;                  // masked pairs -> d = 1
      float d   = sqrtf(d2);
      float ri  = 1.0f / d;                       // d^-1
      float ri2 = 1.0f / d2;                      // d^-2
      s += accs[0][j] * (ri * ri2)                // d^-3
         + accs[1][j] * ri2
         + accs[2][j] * ri
         + accs[3][j] * d
         + accs[4][j] * d2;
    }
    usum[t] = s;
  }

#pragma unroll
  for (int t = 0; t < Tn; ++t) {
    float s = usum[t];
#pragma unroll
    for (int off = 16; off > 0; off >>= 1) s += __shfl_xor(s, off, 32);
    if (lane == 0) partial[((b * Tn + t) * LT + lt) * RT + rt] = s;
  }
}

// ---- Phase 3: deterministic reduction of partials ----------------------
__global__ void reduce_kernel(const float* __restrict__ partial,  // [B*T, 512]
                              float* __restrict__ out) {          // [B*T]
  const int bt   = blockIdx.x;    // 0 .. B*T-1
  const int lane = threadIdx.x;   // 0 .. 31
  const float* p = partial + bt * (LT * RT);
  float s = 0.0f;
#pragma unroll
  for (int k = 0; k < (LT * RT) / 32; ++k) s += p[lane + k * 32];
#pragma unroll
  for (int off = 16; off > 0; off >>= 1) s += __shfl_xor(s, off, 32);
  if (lane == 0) out[bt] = s;
}

// ---- Host launcher ------------------------------------------------------
extern "C" void kernel_launch(void* const* d_in, const int* in_sizes, int n_in,
                              void* d_out, int out_size, void* d_ws, size_t ws_size,
                              hipStream_t stream) {
  const float* lig_feat   = (const float*)d_in[0];
  const float* rec_feat   = (const float*)d_in[1];
  const float* lig_coord  = (const float*)d_in[2];
  const float* rec_coord  = (const float*)d_in[3];
  const float* pre_rot    = (const float*)d_in[4];
  const float* trans      = (const float*)d_in[5];
  const int*   lig_counts = (const int*)d_in[6];
  const int*   rec_counts = (const int*)d_in[7];
  float* out = (float*)d_out;

  // Workspace layout:
  //   new_lig  : B*T*L*3 floats           (196,608 B)
  //   partial  : B*T*LT*RT floats         (262,144 B)
  //   lig_hi/lo: B*L*E*F bf16 each        (5.24 MB each)
  //   rec_hi/lo: B*R*E*F bf16 each        (41.9 MB each)
  constexpr size_t nLig = (size_t)Bn * Ln * En * Fn;   // 2,621,440
  constexpr size_t nRec = (size_t)Bn * Rn * En * Fn;   // 20,971,520
  char* w = (char*)d_ws;
  float*    ws_new_lig = (float*)w;                 w += (size_t)Bn * Tn * Ln * 3 * 4;
  float*    ws_partial = (float*)w;                 w += (size_t)Bn * Tn * LT * RT * 4;
  uint16_t* ws_lig_hi  = (uint16_t*)w;              w += nLig * 2;
  uint16_t* ws_lig_lo  = (uint16_t*)w;              w += nLig * 2;
  uint16_t* ws_rec_hi  = (uint16_t*)w;              w += nRec * 2;
  uint16_t* ws_rec_lo  = (uint16_t*)w;

  pack_kernel<<<2048, 256, 0, stream>>>(lig_feat, ws_lig_hi, ws_lig_lo, (int)nLig);
  pack_kernel<<<4096, 256, 0, stream>>>(rec_feat, ws_rec_hi, ws_rec_lo, (int)nRec);

  prep_kernel<<<Bn * Tn, Ln, 0, stream>>>(lig_coord, pre_rot, trans, ws_new_lig);

  main_kernel<<<Bn * LT * (RT / 8), 256, 0, stream>>>(
      ws_lig_hi, ws_lig_lo, ws_rec_hi, ws_rec_lo,
      rec_coord, lig_counts, rec_counts, ws_new_lig, ws_partial);

  reduce_kernel<<<Bn * Tn, 32, 0, stream>>>(ws_partial, out);
}
